// SpectralConv2d_fast_128849019121
// MI455X (gfx1250) — compile-verified, matrix operand order checked by reading, not run on hardware
//
#include <hip/hip_runtime.h>
#include <hip/hip_bf16.h>

typedef __attribute__((ext_vector_type(16))) _Float16 v16h;
typedef __attribute__((ext_vector_type(8)))  _Float16 v8h;
typedef __attribute__((ext_vector_type(8)))  float    v8f;

#define WMMA16(a, b, c) __builtin_amdgcn_wmma_f32_16x16x32_f16( \
    false, (a), false, (b), (short)0, (c), false, false)

#define TWO_PI_256 0.024543692606170f  // 2*pi/256

// ---------- WMMA fragment helpers (ISA 7.12.2 layouts, wave32) ----------
// A fragment (16x32 f16): lane L holds row (L&15); kb = kBase + (L>>4)*8;
// elems j<8 -> k=kb+j, j>=8 -> k=kb+16+j-8.  Two aligned 16B LDS reads.
// B fragments use the SAME pattern on an N-major (transposed) array:
// lane L holds column (L&15), K runs contiguously.
__device__ __forceinline__ v16h ldsLoadFrag(const _Float16* M, int ld, int majorBase, int kBase) {
  const int lane  = threadIdx.x & 31;
  const int major = majorBase + (lane & 15);
  const int kb    = kBase + ((lane >> 4) << 3);
  const v8h lo = *(const v8h*)(M + major * ld + kb);
  const v8h hi = *(const v8h*)(M + major * ld + kb + 16);
  v16h a;
#pragma unroll
  for (int j = 0; j < 8; ++j) { a[j] = lo[j]; a[j + 8] = hi[j]; }
  return a;
}

__device__ __forceinline__ v16h ldsLoadFragNeg(const _Float16* M, int ld, int majorBase,
                                               int kBase, bool neg) {
  v16h b = ldsLoadFrag(M, ld, majorBase, kBase);
  if (neg) {
#pragma unroll
    for (int j = 0; j < 16; ++j) b[j] = -b[j];
  }
  return b;
}

// D (16x16 f32): lane L holds col (L&15); vgpr r -> row r + (L>>4)*8.
// Store into a row-major f16 array (column write, strided).
__device__ __forceinline__ void storeD16(_Float16* M, int ld, int rowBase, int colBase, v8f acc) {
  const int lane = threadIdx.x & 31;
  const int col  = colBase + (lane & 15);
  const int r0   = rowBase + ((lane >> 4) << 3);
#pragma unroll
  for (int r = 0; r < 8; ++r) M[(r0 + r) * ld + col] = (_Float16)acc[r];
}

// Store D into a TRANSPOSED (N-major) f16 array: one contiguous 16B write.
__device__ __forceinline__ void storeD16T(_Float16* M, int ld, int rowBase, int colBase, v8f acc) {
  const int lane = threadIdx.x & 31;
  const int col  = colBase + (lane & 15);
  const int r0   = rowBase + ((lane >> 4) << 3);
  v8h hv;
#pragma unroll
  for (int r = 0; r < 8; ++r) hv[r] = (_Float16)acc[r];
  *(v8h*)(M + col * ld + r0) = hv;
}

// =========================================================================
// Kernel 1: forward truncated DFT.  One block per (b, cin) image.
//   Stage A: Fw(256x16, cplx) = X(256x256) @ Dw(256x16, cplx)   [K streamed]
//   Stage B: G(32x16, cplx)   = Dh(32x256, cplx) @ Fw           [packed K=512]
// Output g layout: [ky(32)][kx(16)][part(2)][cin(64)][b(16)]  (f32)
// =========================================================================
__global__ __launch_bounds__(256) void k_fwd(const float* __restrict__ x,
                                             float* __restrict__ g) {
  __shared__ _Float16 Xs[256 * 72];        // 36 KB  X chunk (f16), A operand
  __shared__ _Float16 DwT[2][16 * 72];     // 4.5 KB W-dir twiddles, transposed [n][k]
  __shared__ _Float16 Ah[2][32 * 512];     // 64 KB  packed H-dir DFT rows (A operand)
  __shared__ _Float16 FwsT[16 * 520];      // 16 KB  packed [Fwr;Fwi], transposed [n][kk]

  const int img  = blockIdx.x;
  const int bb   = img >> 6;
  const int ci   = img & 63;
  const int tid  = threadIdx.x;
  const int wv   = tid >> 5;
  const int lane = tid & 31;

  // Packed H-direction DFT operands. e^{-i t}: Dhr=cos t, Dhi=-sin t.
  // Ah[0] = [Dhr | -Dhi] = [c | s],  Ah[1] = [Dhi | Dhr] = [-s | c]
  for (int idx = tid; idx < 32 * 512; idx += 256) {
    int j  = idx >> 9;
    int kk = idx & 511;
    int h  = kk & 255;
    int hi = kk >> 8;
    int hy = (j < 16) ? j : (224 + j);        // 240..255 for j=16..31
    float t = (float)((hy * h) & 255) * TWO_PI_256;
    float s, c;
    __sincosf(t, &s, &c);
    Ah[0][idx] = (_Float16)(hi ? s : c);
    Ah[1][idx] = (_Float16)(hi ? c : -s);
  }

  v8f zero = {0.f, 0.f, 0.f, 0.f, 0.f, 0.f, 0.f, 0.f};
  v8f acc[4];
#pragma unroll
  for (int q = 0; q < 4; ++q) acc[q] = zero;

  // Stream K (the W axis) in chunks of 64.
  for (int k0 = 0; k0 < 256; k0 += 64) {
    __syncthreads();
    // Load X chunk (coalesced: 16 lanes cover a 256B row segment).
    {
      const int cg = tid & 15, rg = tid >> 4;
#pragma unroll
      for (int p = 0; p < 16; ++p) {
        int h = p * 16 + rg;
        float4 v = *(const float4*)(x + (size_t)img * 65536 + (size_t)h * 256 + k0 + cg * 4);
        _Float16* dst = &Xs[h * 72 + cg * 4];
        dst[0] = (_Float16)v.x; dst[1] = (_Float16)v.y;
        dst[2] = (_Float16)v.z; dst[3] = (_Float16)v.w;
      }
    }
    // W-direction twiddle chunk, stored transposed: DwT[part][n][k].
    for (int idx = tid; idx < 1024; idx += 256) {
      int k = idx >> 4, n = idx & 15;
      float t = (float)(((k0 + k) * n) & 255) * TWO_PI_256;
      float s, c;
      __sincosf(t, &s, &c);
      DwT[0][n * 72 + k] = (_Float16)c;
      DwT[1][n * 72 + k] = (_Float16)(-s);
    }
    __syncthreads();
    // Stage A accumulate: wave wv owns row-tiles 2*wv, 2*wv+1 (re & im).
#pragma unroll
    for (int kt = 0; kt < 2; ++kt) {
      v16h br = ldsLoadFrag(DwT[0], 72, 0, kt * 32);
      v16h bi = ldsLoadFrag(DwT[1], 72, 0, kt * 32);
#pragma unroll
      for (int mq = 0; mq < 2; ++mq) {
        int mt = wv * 2 + mq;
        v16h a = ldsLoadFrag(Xs, 72, mt * 16, kt * 32);
        acc[2 * mq]     = WMMA16(a, br, acc[2 * mq]);
        acc[2 * mq + 1] = WMMA16(a, bi, acc[2 * mq + 1]);
      }
    }
  }

  // Spill Fw into packed transposed [n][kk]: kk<256 -> Fwr, kk>=256 -> Fwi.
#pragma unroll
  for (int q = 0; q < 4; ++q) {
    int mt = wv * 2 + (q >> 1);
    int part = q & 1;
    storeD16T(FwsT, 520, part * 256 + mt * 16, 0, acc[q]);
  }
  __syncthreads();

  // Stage B: 4 tiles (2 ky-tiles x re/im), waves 0..3, K = 512.
  if (wv < 4) {
    const int mt2 = wv >> 1, part = wv & 1;
    v8f gacc = zero;
#pragma unroll
    for (int kt = 0; kt < 16; ++kt) {
      v16h a = ldsLoadFrag(Ah[part], 512, mt2 * 16, kt * 32);
      v16h b = ldsLoadFrag(FwsT, 520, 0, kt * 32);
      gacc = WMMA16(a, b, gacc);
    }
    const int n  = lane & 15;
    const int r0 = mt2 * 16 + ((lane >> 4) << 3);
#pragma unroll
    for (int r = 0; r < 8; ++r) {
      int j = r0 + r;
      g[(((size_t)(j * 16 + n) * 2 + part) * 64 + ci) * 16 + bb] = gacc[r];
    }
  }
}

// =========================================================================
// Kernel 2: per-mode complex channel mix.  One block per mode (32*16 = 512).
//   out[b,o] = sum_i X[b,i] * W[i,o]   (complex), M=16 N=64 K=64 -> packed K=128
// Output g2 layout: [b(16)][cout(64)][ky(32)][kx(16)][part(2)]  (f32)
// =========================================================================
__global__ __launch_bounds__(256) void k_mix(const float* __restrict__ g,
                                             const float* __restrict__ w1r,
                                             const float* __restrict__ w1i,
                                             const float* __restrict__ w2r,
                                             const float* __restrict__ w2i,
                                             float* __restrict__ g2) {
  __shared__ _Float16 As[16 * 136];   // packed A = [Xr | Xi], 16 x 128, ld=136
  __shared__ _Float16 WrT[64 * 72];   // 9 KB  transposed [o][i]
  __shared__ _Float16 WiT[64 * 72];   // 9 KB  transposed [o][i]

  const int mode = blockIdx.x;
  const int j = mode >> 4, n = mode & 15;
  const int mm = (j < 16) ? j : (j - 16);
  const float* wr = (j < 16) ? w1r : w2r;
  const float* wi = (j < 16) ? w1i : w2i;
  const int tid = threadIdx.x, wv = tid >> 5, lane = tid & 31;

  // A operand: ws1 slice for this mode is contiguous [part][i][b] = 2048 floats.
  {
    const size_t base = (size_t)(j * 16 + n) * 2048;
    for (int idx = tid; idx < 2048; idx += 256) {
      int kk = idx >> 4, bi = idx & 15;
      As[bi * 136 + kk] = (_Float16)g[base + idx];
    }
  }
  // Weights for this mode: element (i,o) at (i*64+o)*256 + mm*16 + n.
  for (int idx = tid; idx < 4096; idx += 256) {
    int i = idx >> 6, o = idx & 63;
    size_t off = (size_t)idx * 256 + (size_t)mm * 16 + n;
    WrT[o * 72 + i] = (_Float16)wr[off];
    WiT[o * 72 + i] = (_Float16)wi[off];
  }
  __syncthreads();

  // out_r = [Xr|Xi] @ [Wr; -Wi],  out_i = [Xr|Xi] @ [Wi; Wr]
  const int part = wv >> 2, ct = wv & 3;
  const _Float16* m0 = part ? WiT : WrT;
  const _Float16* m1 = part ? WrT : WiT;
  const bool neg = (part == 0);
  v8f zero = {0.f, 0.f, 0.f, 0.f, 0.f, 0.f, 0.f, 0.f};
  v8f acc = zero;
#pragma unroll
  for (int kt = 0; kt < 4; ++kt) {
    v16h a = ldsLoadFrag(As, 136, 0, kt * 32);
    v16h b = (kt < 2) ? ldsLoadFrag(m0, 72, ct * 16, kt * 32)
                      : ldsLoadFragNeg(m1, 72, ct * 16, (kt - 2) * 32, neg);
    acc = WMMA16(a, b, acc);
  }
  const int o  = ct * 16 + (lane & 15);
  const int r0 = (lane >> 4) << 3;
#pragma unroll
  for (int r = 0; r < 8; ++r) {
    int b = r0 + r;
    g2[(((size_t)(b * 64 + o) * 32 + j) * 16 + n) * 2 + part] = acc[r];
  }
}

// =========================================================================
// Kernel 3: inverse truncated DFT.  One block per (b, cout) image.
//   Stage C: T(256x16, cplx) = iDh(256x32, cplx) @ G2(32x16, cplx)  [K=64 packed]
//   Stage D: out(256x256)    = [Tr | Ti](256x32) @ E(32x256)        [K=32]
// Scale 1/65536 split as (1/256) in Ac and (ck/256) in E to stay f16-normal.
// =========================================================================
__global__ __launch_bounds__(256) void k_inv(const float* __restrict__ g2,
                                             float* __restrict__ out) {
  __shared__ _Float16 Ac[2][256 * 64];  // 64 KB  [iDhr|-iDhi], [iDhi|iDhr] (x 1/256)
  __shared__ _Float16 BcT[16 * 72];     // 2.3 KB packed [G2r; G2i], transposed [n][kk]
  __shared__ _Float16 Tp[256 * 32];     // 16 KB  packed [Tr | Ti] (A operand)
  __shared__ _Float16 EsT[256 * 40];    // 20 KB  final iDFT operand, transposed [w][kk2]

  const int img = blockIdx.x;
  const int tid = threadIdx.x, wv = tid >> 5, lane = tid & 31;

  // iDFT over ky: e^{+i t}: cr=cos, ci=sin.  Ac[0]=[c|-s]/256, Ac[1]=[s|c]/256.
  for (int idx = tid; idx < 16384; idx += 256) {
    int h = idx >> 6, kk = idx & 63;
    int j = kk & 31, half = kk >> 5;
    int hy = (j < 16) ? j : (224 + j);
    float t = (float)((hy * h) & 255) * TWO_PI_256;
    float s, c;
    __sincosf(t, &s, &c);
    Ac[0][idx] = (_Float16)((half ? -s : c) * 0.00390625f);
    Ac[1][idx] = (_Float16)((half ? c : s) * 0.00390625f);
  }
  // E transposed [w][kk2]: kk2<16 pairs Tr (ck*cos/256), kk2>=16 pairs Ti (-ck*sin/256).
  for (int idx = tid; idx < 8192; idx += 256) {
    int kk2 = idx >> 8, w = idx & 255;
    int k = kk2 & 15, half = kk2 >> 4;
    float ck = (k == 0) ? 0.00390625f : 0.0078125f;
    float t = (float)((k * w) & 255) * TWO_PI_256;
    float s, c;
    __sincosf(t, &s, &c);
    EsT[w * 40 + kk2] = (_Float16)((half ? -s : c) * ck);
  }
  // Modes for this (b, cout): contiguous 1024 floats -> BcT[n][part*32+j].
  {
    const float* src = g2 + (size_t)img * 1024;
    for (int idx = tid; idx < 1024; idx += 256) {
      int j = idx >> 5, n = (idx >> 1) & 15, part = idx & 1;
      BcT[n * 72 + part * 32 + j] = (_Float16)src[idx];
    }
  }
  __syncthreads();

  v8f zero = {0.f, 0.f, 0.f, 0.f, 0.f, 0.f, 0.f, 0.f};

  // Stage C: 32 tiles (16 row-tiles x re/im), 4 per wave, K = 64.
#pragma unroll
  for (int q = 0; q < 4; ++q) {
    int t = wv * 4 + q;
    int mt = t >> 1, part = t & 1;
    v8f acc = zero;
#pragma unroll
    for (int kt = 0; kt < 2; ++kt) {
      v16h a = ldsLoadFrag(Ac[part], 64, mt * 16, kt * 32);
      v16h b = ldsLoadFrag(BcT, 72, 0, kt * 32);
      acc = WMMA16(a, b, acc);
    }
    storeD16(Tp, 32, mt * 16, part * 16, acc);  // [Tr | Ti] packed along K
  }
  __syncthreads();

  // Stage D: 256 output tiles, 32 per wave, one WMMA each; D goes to global.
#pragma unroll
  for (int mq = 0; mq < 2; ++mq) {
    int th = wv * 2 + mq;
    v16h a = ldsLoadFrag(Tp, 32, th * 16, 0);
    for (int tw = 0; tw < 16; ++tw) {
      v16h b = ldsLoadFrag(EsT, 40, tw * 16, 0);
      v8f d = WMMA16(a, b, zero);
      const int w  = tw * 16 + (lane & 15);
      const int h0 = th * 16 + ((lane >> 4) << 3);
#pragma unroll
      for (int r = 0; r < 8; ++r)
        out[(size_t)img * 65536 + (size_t)(h0 + r) * 256 + w] = d[r];
    }
  }
}

// =========================================================================
extern "C" void kernel_launch(void* const* d_in, const int* in_sizes, int n_in,
                              void* d_out, int out_size, void* d_ws, size_t ws_size,
                              hipStream_t stream) {
  const float* x   = (const float*)d_in[0];
  const float* w1r = (const float*)d_in[1];
  const float* w1i = (const float*)d_in[2];
  const float* w2r = (const float*)d_in[3];
  const float* w2i = (const float*)d_in[4];
  float* out = (float*)d_out;

  float* ws1 = (float*)d_ws;            // 4 MB: modes of x_ft
  float* ws2 = ws1 + (1u << 20);        // 4 MB: mixed modes

  k_fwd<<<1024, 256, 0, stream>>>(x, ws1);
  k_mix<<<512, 256, 0, stream>>>(ws1, w1r, w1i, w2r, w2i, ws2);
  k_inv<<<1024, 256, 0, stream>>>(ws2, out);
}
